// MarketTranformer_75746043232766
// MI455X (gfx1250) — matrix operand order repaired and checked by reading, not compiled, not run on hardware
//
#include <hip/hip_runtime.h>
#include <hip/hip_bf16.h>
#include <math.h>

typedef __attribute__((ext_vector_type(16))) _Float16 v16h;
typedef __attribute__((ext_vector_type(8)))  float    v8f;

#define NN  4096
#define EE  131072
#define DD  64
#define HH  8
#define DHH 8
#define FFD 2048
#define CC  3

static __device__ __forceinline__ v8f wmma_f16(v16h a, v16h b, v8f c) {
  // D = A(16x32 f16) * B(32x16 f16) + C(16x16 f32)
  return __builtin_amdgcn_wmma_f32_16x16x32_f16(false, a, false, b, (short)0, c,
                                                false, false);
}

// ---------------------------------------------------------------------------
// Strip WMMA GEMM: each wave computes a 16x64 strip of C = act(A@W + bias +resid)
// 4 accumulators share one A fragment -> 4 back-to-back WMMAs per 32-K chunk.
// BIAS/RES/RELU are compile-time so the epilogue is branch-free.
// ---------------------------------------------------------------------------
template <bool BIAS, bool RES, bool RELU>
__global__ void gemm_strip_kernel(const float* __restrict__ A, const float* __restrict__ W,
                                  const float* __restrict__ bias, const float* __restrict__ resid,
                                  float* __restrict__ C, int M, int N, int K) {
  const int lane = threadIdx.x & 31;
  const int wave = blockIdx.x * (blockDim.x >> 5) + (threadIdx.x >> 5);
  const int nstrips = N >> 6;                      // 64 output cols per strip
  if (wave >= (M >> 4) * nstrips) return;          // wave-uniform guard
  const int m0 = (wave / nstrips) << 4;
  const int n0 = (wave % nstrips) << 6;
  const int half = lane >> 4;
  const int lc = lane & 15;
  const int arow = m0 + lc;

  v8f acc[4] = {{}, {}, {}, {}};
  for (int k0 = 0; k0 < K; k0 += 32) {
    if (k0 + 32 < K)
      __builtin_prefetch(&A[arow * K + k0 + 32], 0, 0);   // global_prefetch_b8
    v16h af;
#pragma unroll
    for (int i = 0; i < 16; ++i) {                 // A 16x32: lane=row, halves cover K
      int kk = (half ? 16 : 0) + i;                // 0-15 lo half, 16-31 hi half
      af[i] = (_Float16)A[arow * K + k0 + kk];
    }
    const int kb = k0 + (half ? 16 : 0);           // B: lanes 0-15 K=0-15, 16-31 K=16-31
#pragma unroll
    for (int t = 0; t < 4; ++t) {
      v16h bf;
      const int bcol = n0 + (t << 4) + lc;
#pragma unroll
      for (int i = 0; i < 16; ++i)
        bf[i] = (_Float16)W[(kb + i) * N + bcol];
      acc[t] = wmma_f16(af, bf, acc[t]);
    }
  }

  const int mb = m0 + (half << 3);                 // C layout: VGPR r -> row mb+r
#pragma unroll
  for (int t = 0; t < 4; ++t) {
    const int col = n0 + (t << 4) + lc;
#pragma unroll
    for (int r = 0; r < 8; ++r) {
      float vv = acc[t][r];
      if (BIAS) vv += bias[col];
      if (RES)  vv += resid[(mb + r) * N + col];
      if (RELU) vv = fmaxf(vv, 0.0f);
      C[(mb + r) * N + col] = vv;
    }
  }
}

// ---------------------------------------------------------------------------
// Fused flash attention: wave = (head, 16-query tile); keys in chunks of 32.
// Score WMMA K-dim = head_dim(8) zero-padded to 32; online softmax; P via LDS.
// All fragment loads are unconditional (clamped addresses); zero-padding is
// done with register selects so EXEC never diverges around memory ops.
// ---------------------------------------------------------------------------
__global__ void flash_attn_kernel(const float* __restrict__ q, const float* __restrict__ k,
                                  const float* __restrict__ v, float* __restrict__ o) {
  __shared__ _Float16 plds[8][16 * 32];            // per-wave 16x32 P tile
  const int lane = threadIdx.x & 31;
  const int wv = threadIdx.x >> 5;
  const int wave = blockIdx.x * (blockDim.x >> 5) + wv;
  const int head = wave >> 8;                      // 8 heads x 256 query tiles
  const int qt = wave & 255;
  if (head >= HH) return;
  const int half = lane >> 4;
  const bool lo = (half == 0);
  const int lc = lane & 15;
  const int lcc = lc & 7;                          // clamped dim index
  const int qrow = (qt << 4) + lc;

  // Q fragment: 8 contiguous loads per lane; pad via selects (no exec divergence)
  float qv[8];
#pragma unroll
  for (int i = 0; i < 8; ++i) qv[i] = q[qrow * DD + head * DHH + i];
  v16h af;
#pragma unroll
  for (int i = 0; i < 8; ++i) af[i] = lo ? (_Float16)qv[i] : (_Float16)0.0f;
#pragma unroll
  for (int i = 8; i < 16; ++i) af[i] = (_Float16)0.0f;

  float mrow[8], lrow[8];
  v8f oacc = {};
#pragma unroll
  for (int r = 0; r < 8; ++r) { mrow[r] = -1e30f; lrow[r] = 0.0f; }
  const float sc = 0.35355339059327373f;           // 1/sqrt(8)

  for (int kt = 0; kt < NN; kt += 32) {
    // K chunk: per-lane contiguous 8-float rows, padded via selects
    float k0v[8], k1v[8];
#pragma unroll
    for (int i = 0; i < 8; ++i) {
      k0v[i] = k[(kt + lc) * DD + head * DHH + i];
      k1v[i] = k[(kt + 16 + lc) * DD + head * DHH + i];
    }
    v16h b0, b1;
#pragma unroll
    for (int i = 0; i < 8; ++i) {
      b0[i] = lo ? (_Float16)k0v[i] : (_Float16)0.0f;
      b1[i] = lo ? (_Float16)k1v[i] : (_Float16)0.0f;
    }
#pragma unroll
    for (int i = 8; i < 16; ++i) { b0[i] = (_Float16)0.0f; b1[i] = (_Float16)0.0f; }

    v8f z = {};
    v8f s0 = wmma_f16(af, b0, z);
    v8f s1 = wmma_f16(af, b1, z);

#pragma unroll
    for (int r = 0; r < 8; ++r) {                  // online softmax, row = 8*half+r
      float a0 = s0[r] * sc, a1 = s1[r] * sc;
      float rowmax = fmaxf(a0, a1);
#pragma unroll
      for (int m = 8; m > 0; m >>= 1) rowmax = fmaxf(rowmax, __shfl_xor(rowmax, m, 32));
      float newm = fmaxf(mrow[r], rowmax);
      float scale = __expf(mrow[r] - newm);
      float e0 = __expf(a0 - newm);
      float e1 = __expf(a1 - newm);
      float rs = e0 + e1;
#pragma unroll
      for (int m = 8; m > 0; m >>= 1) rs += __shfl_xor(rs, m, 32);
      lrow[r] = lrow[r] * scale + rs;
      mrow[r] = newm;
      oacc[r] *= scale;
      int prow = (half << 3) + r;                  // stage P (C-layout -> LDS row-major)
      plds[wv][prow * 32 + lc] = (_Float16)e0;
      plds[wv][prow * 32 + 16 + lc] = (_Float16)e1;
    }
    asm volatile("s_wait_dscnt 0" ::: "memory");

    v16h pf;                                       // reload P in A-fragment layout
#pragma unroll
    for (int i = 0; i < 16; ++i) {
      int kk = (half ? 16 : 0) + i;
      pf[i] = plds[wv][lc * 32 + kk];
    }
    // V chunk: clamped-column unconditional loads + selects
    v16h bv;
#pragma unroll
    for (int i = 0; i < 16; ++i) {
      int kr = (half << 4) + i;
      float vv = v[(kt + kr) * DD + head * DHH + lcc];
      bv[i] = (lc < DHH) ? (_Float16)vv : (_Float16)0.0f;
    }
    oacc = wmma_f16(pf, bv, oacc);
  }

  if (lc < DHH) {
#pragma unroll
    for (int r = 0; r < 8; ++r) {
      int row = (qt << 4) + (half << 3) + r;
      o[row * DD + head * DHH + lc] = oacc[r] / lrow[r];
    }
  }
}

// ---------------------------------------------------------------------------
// LayerNorm over D=64, one wave per row (2 elements per lane), in place.
// ---------------------------------------------------------------------------
__global__ void layernorm_kernel(float* __restrict__ x, const float* __restrict__ g,
                                 const float* __restrict__ b) {
  const int lane = threadIdx.x & 31;
  const int row = blockIdx.x * (blockDim.x >> 5) + (threadIdx.x >> 5);
  if (row >= NN) return;
  float x0 = x[row * DD + lane];
  float x1 = x[row * DD + 32 + lane];
  float s = x0 + x1;
#pragma unroll
  for (int m = 16; m > 0; m >>= 1) s += __shfl_xor(s, m, 32);
  float mu = s * (1.0f / 64.0f);
  float d0 = x0 - mu, d1 = x1 - mu;
  float vs = d0 * d0 + d1 * d1;
#pragma unroll
  for (int m = 16; m > 0; m >>= 1) vs += __shfl_xor(vs, m, 32);
  float rstd = rsqrtf(vs * (1.0f / 64.0f) + 1e-5f);
  x[row * DD + lane]      = d0 * rstd * g[lane] + b[lane];
  x[row * DD + 32 + lane] = d1 * rstd * g[32 + lane] + b[32 + lane];
}

// ---------------------------------------------------------------------------
// Embedding (4->64) + sinusoidal positional encoding
// ---------------------------------------------------------------------------
__global__ void embed_pe_kernel(const float* __restrict__ x, const float* __restrict__ W,
                                const float* __restrict__ b, float* __restrict__ h) {
  int tid = blockIdx.x * blockDim.x + threadIdx.x;
  if (tid >= NN * DD) return;
  int n = tid >> 6, d = tid & 63;
  float acc = b[d];
#pragma unroll
  for (int i = 0; i < 4; ++i) acc += x[n * 4 + i] * W[i * DD + d];
  int i2 = d >> 1;
  float div = __expf((float)(2 * i2) * (-9.210340371976184f / 64.0f)); // -ln(1e4)/D
  float ang = (float)n * div;
  acc += (d & 1) ? __cosf(ang) : __sinf(ang);
  h[tid] = acc;
}

// ---------------------------------------------------------------------------
// GATv2 scatter pipeline
// ---------------------------------------------------------------------------
static __device__ __forceinline__ unsigned fenc(float f) {   // monotone f32 -> u32
  unsigned u = __float_as_uint(f);
  return (u & 0x80000000u) ? ~u : (u | 0x80000000u);
}
static __device__ __forceinline__ float fdec(unsigned u) {
  return (u & 0x80000000u) ? __uint_as_float(u & 0x7fffffffu) : __uint_as_float(~u);
}

__global__ void gat_init_kernel(unsigned* __restrict__ emax, float* __restrict__ den) {
  int tid = blockIdx.x * blockDim.x + threadIdx.x;
  if (tid >= NN * HH) return;
  emax[tid] = fenc(-INFINITY);
  den[tid] = 0.0f;
}

__global__ void gat_out_init_kernel(float* __restrict__ out, const float* __restrict__ b) {
  int tid = blockIdx.x * blockDim.x + threadIdx.x;
  if (tid >= NN * DD) return;
  out[tid] = b[tid & 63];
}

__global__ void gat_edge_score_kernel(const int* __restrict__ eidx, const float* __restrict__ xl,
                                      const float* __restrict__ xr, const float* __restrict__ att,
                                      float* __restrict__ esc, unsigned* __restrict__ emax) {
  int tid = blockIdx.x * blockDim.x + threadIdx.x;
  if (tid >= EE * HH) return;
  int e = tid >> 3, hh = tid & 7;
  int src = eidx[e], dst = eidx[EE + e];
  float acc = 0.0f;
#pragma unroll
  for (int d = 0; d < DHH; ++d) {
    float mv = xl[src * DD + hh * DHH + d] + xr[dst * DD + hh * DHH + d];
    mv = (mv > 0.0f) ? mv : 0.2f * mv;           // leaky_relu(0.2)
    acc += mv * att[hh * DHH + d];
  }
  esc[tid] = acc;
  atomicMax(&emax[dst * HH + hh], fenc(acc));
}

__global__ void gat_edge_w_kernel(const int* __restrict__ eidx, const float* __restrict__ esc,
                                  const unsigned* __restrict__ emax, float* __restrict__ wbuf,
                                  float* __restrict__ den) {
  int tid = blockIdx.x * blockDim.x + threadIdx.x;
  if (tid >= EE * HH) return;
  int e = tid >> 3, hh = tid & 7;
  int dst = eidx[EE + e];
  float w = __expf(esc[tid] - fdec(emax[dst * HH + hh]));
  wbuf[tid] = w;
  atomicAdd(&den[dst * HH + hh], w);
}

__global__ void gat_scatter_kernel(const int* __restrict__ eidx, const float* __restrict__ wbuf,
                                   const float* __restrict__ den, const float* __restrict__ xl,
                                   float* __restrict__ out) {
  int tid = blockIdx.x * blockDim.x + threadIdx.x;
  if (tid >= EE * HH) return;
  int e = tid >> 3, hh = tid & 7;
  int src = eidx[e], dst = eidx[EE + e];
  float alpha = wbuf[tid] / (den[dst * HH + hh] + 1e-16f);
#pragma unroll
  for (int d = 0; d < DHH; ++d)
    atomicAdd(&out[dst * DD + hh * DHH + d], alpha * xl[src * DD + hh * DHH + d]);
}

// ---------------------------------------------------------------------------
// Classifier: out[N,3] = h @ clsW + clsB
// ---------------------------------------------------------------------------
__global__ void cls_kernel(const float* __restrict__ h, const float* __restrict__ W,
                           const float* __restrict__ b, float* __restrict__ out) {
  int tid = blockIdx.x * blockDim.x + threadIdx.x;
  if (tid >= NN * CC) return;
  int n = tid / CC, c = tid % CC;
  float acc = b[c];
#pragma unroll
  for (int d = 0; d < DD; ++d) acc += h[n * DD + d] * W[d * CC + c];
  out[tid] = acc;
}

// ---------------------------------------------------------------------------
extern "C" void kernel_launch(void* const* d_in, const int* in_sizes, int n_in,
                              void* d_out, int out_size, void* d_ws, size_t ws_size,
                              hipStream_t stream) {
  (void)in_sizes; (void)n_in; (void)out_size; (void)ws_size;
  // input order: x, edge_index, emb(2), attn0(8), gat 2x4, enc 2x16, cls(2)
  const float* x    = (const float*)d_in[0];
  const int*   eidx = (const int*)d_in[1];
  const float* embW = (const float*)d_in[2];
  const float* embB = (const float*)d_in[3];
  const float* clsW = (const float*)d_in[52];
  const float* clsB = (const float*)d_in[53];

  float* ws = (float*)d_ws;
  float* fH   = ws;                  // [N,D] ping
  float* fA   = fH + NN * DD;        // attn-out / xl temp
  float* fQ   = fA + NN * DD;
  float* fK   = fQ + NN * DD;
  float* fV   = fK + NN * DD;
  float* fB   = fV + NN * DD;        // [N,D] pong
  float* fBig = fB + NN * DD;        // [N,FF], reused for GAT edge buffers
  unsigned* uMax = (unsigned*)(fBig + NN * FFD);
  float* fDen = (float*)(uMax + NN * HH);

  auto gemm = [&](const float* A, const float* W, const float* bias, const float* resid,
                  float* C, int M, int N, int K, int relu) {
    int strips = (M / 16) * (N / 64);
    int blocks = (strips + 7) / 8;                       // 8 waves / 256-thread block
    if (bias && resid)
      gemm_strip_kernel<true, true, false><<<blocks, 256, 0, stream>>>(A, W, bias, resid, C, M, N, K);
    else if (bias && relu)
      gemm_strip_kernel<true, false, true><<<blocks, 256, 0, stream>>>(A, W, bias, resid, C, M, N, K);
    else if (bias)
      gemm_strip_kernel<true, false, false><<<blocks, 256, 0, stream>>>(A, W, bias, resid, C, M, N, K);
    else
      gemm_strip_kernel<false, false, false><<<blocks, 256, 0, stream>>>(A, W, bias, resid, C, M, N, K);
  };
  auto mha = [&](const float* in, const float* const* p, const float* resid, float* outp) {
    gemm(in, p[0], p[1], nullptr, fQ, NN, DD, DD, 0);
    gemm(in, p[2], p[3], nullptr, fK, NN, DD, DD, 0);
    gemm(in, p[4], p[5], nullptr, fV, NN, DD, DD, 0);
    flash_attn_kernel<<<256, 256, 0, stream>>>(fQ, fK, fV, fA);   // 2048 waves
    gemm(fA, p[6], p[7], resid, outp, NN, DD, DD, 0);
  };

  // 1) embedding + positional encoding
  embed_pe_kernel<<<(NN * DD) / 256, 256, 0, stream>>>(x, embW, embB, fH);
  float* cur = fH;
  float* nxt = fB;

  // 2) attn0 (no residual, replaces h)
  {
    const float* p[8];
    for (int i = 0; i < 8; ++i) p[i] = (const float*)d_in[4 + i];
    mha(cur, p, nullptr, nxt);
    float* t = cur; cur = nxt; nxt = t;
  }

  // 3) GATv2 layers
  for (int l = 0; l < 2; ++l) {
    const float* Wl  = (const float*)d_in[12 + 4 * l + 0];
    const float* Wr  = (const float*)d_in[12 + 4 * l + 1];
    const float* att = (const float*)d_in[12 + 4 * l + 2];
    const float* gb  = (const float*)d_in[12 + 4 * l + 3];
    gemm(cur, Wl, nullptr, nullptr, fA, NN, DD, DD, 0);   // xl
    gemm(cur, Wr, nullptr, nullptr, fQ, NN, DD, DD, 0);   // xr
    gat_init_kernel<<<(NN * HH + 255) / 256, 256, 0, stream>>>(uMax, fDen);
    gat_out_init_kernel<<<(NN * DD + 255) / 256, 256, 0, stream>>>(nxt, gb);
    float* esc = fBig;
    float* wb  = fBig + EE * HH;
    int eb = (EE * HH + 255) / 256;
    gat_edge_score_kernel<<<eb, 256, 0, stream>>>(eidx, fA, fQ, att, esc, uMax);
    gat_edge_w_kernel<<<eb, 256, 0, stream>>>(eidx, esc, uMax, wb, fDen);
    gat_scatter_kernel<<<eb, 256, 0, stream>>>(eidx, wb, fDen, fA, nxt);
    float* t = cur; cur = nxt; nxt = t;
  }

  // 4) transformer encoder layers
  for (int l = 0; l < 2; ++l) {
    const float* p[16];
    for (int i = 0; i < 16; ++i) p[i] = (const float*)d_in[20 + 16 * l + i];
    mha(cur, p, cur, nxt);                                 // nxt = cur + MHA(cur)
    layernorm_kernel<<<512, 256, 0, stream>>>(nxt, p[8], p[9]);
    { float* t = cur; cur = nxt; nxt = t; }
    gemm(cur, p[10], p[11], nullptr, fBig, NN, FFD, DD, 1); // relu(h@W1+b1)
    gemm(fBig, p[12], p[13], cur, nxt, NN, DD, FFD, 0);     // + residual
    layernorm_kernel<<<512, 256, 0, stream>>>(nxt, p[14], p[15]);
    { float* t = cur; cur = nxt; nxt = t; }
  }

  // 5) classifier
  cls_kernel<<<(NN * CC + 255) / 256, 256, 0, stream>>>(cur, clsW, clsB, (float*)d_out);
}
